// LocalAttention_53790170415657
// MI455X (gfx1250) — compile-verified, matrix operand order checked by reading, not run on hardware
//
#include <hip/hip_runtime.h>

#define SEQ   4096
#define EMB   1024
#define NH    16
#define HD    64
#define WIN   128
#define NKEY  272   // 17*16 keys spanned by a 16-query tile's window union
#define NKPAD 288   // padded to a multiple of 32 for the PV K-loop

typedef __attribute__((ext_vector_type(16))) __bf16 v16bf;
typedef __attribute__((ext_vector_type(8)))  __bf16 v8bf;
typedef __attribute__((ext_vector_type(8)))  float  v8f;

static __device__ __forceinline__ v8bf ld8(const __bf16* p) {
    return *(const v8bf*)p;   // 16-byte load (global or LDS)
}

static __device__ __forceinline__ v16bf mk16(v8bf lo, v8bf hi) {
    v16bf r;
#pragma unroll
    for (int i = 0; i < 8; ++i) { r[i] = lo[i]; r[i + 8] = hi[i]; }
    return r;
}

// Guarded 8-element run load along the key dimension (rowbase[k0..k0+7]),
// zero-filling out-of-range keys.
static __device__ __forceinline__ v8bf ld8_keys(const __bf16* rowbase, int k0, int lim) {
    v8bf r;
    if (k0 >= 0 && k0 + 8 <= lim) {
        r = ld8(rowbase + k0);
    } else {
#pragma unroll
        for (int i = 0; i < 8; ++i) {
            int k = k0 + i;
            r[i] = (k >= 0 && k < lim) ? rowbase[k] : (__bf16)0.0f;
        }
    }
    return r;
}

static __device__ __forceinline__ v8f wmma_bf16(v16bf a, v16bf b, v8f c) {
    // D = A(16x32 bf16) x B(32x16 bf16) + C(16x16 f32)
    return __builtin_amdgcn_wmma_f32_16x16x32_bf16(false, a, false, b, (short)0, c,
                                                   false, false);
}

// ---------------------------------------------------------------------------
// Stage 0: conversions
// ---------------------------------------------------------------------------
__global__ void k_cvt_bf16(const float* __restrict__ in, __bf16* __restrict__ out, int n) {
    int i = blockIdx.x * blockDim.x + threadIdx.x;
    if (i < n) out[i] = (__bf16)in[i];
}

// W is [K=EMB][N=EMB] row-major; write Wt[n][k] (bf16) so the WMMA B operand
// (columns of W) reads contiguous 16-byte runs along K.
__global__ void k_cvt_transpose(const float* __restrict__ w, __bf16* __restrict__ wt) {
    int t = blockIdx.x * blockDim.x + threadIdx.x;   // E*E threads
    int k = t >> 10;
    int n = t & (EMB - 1);
    wt[(size_t)n * EMB + k] = (__bf16)w[t];
}

// ---------------------------------------------------------------------------
// Stage 1/3: bf16 GEMM. One wave computes a 32x64 output tile
// (2 M-subtiles x 4 N-subtiles, 8 f32 accumulators) so each A tile is
// reused 4x and each B tile 2x: ~21 FLOP/byte from the cache hierarchy.
//  A  : [M x EMB] bf16 row-major
//  Bt : [N x EMB] bf16 (transposed weights: Bt[n][k] = W[k][n])
//  mode 1: out bf16 in [B,H,S,D] layout (Q,K)
//  mode 2: out bf16 in [B,H,D,S] layout (V transposed)
//  mode 3: out f32  [M x N] (final projection -> d_out)
// ---------------------------------------------------------------------------
__global__ void __launch_bounds__(256)
k_gemm(const __bf16* __restrict__ A, const __bf16* __restrict__ Bt,
       const float* __restrict__ bias, void* __restrict__ outp, int mode) {
    const int lane = threadIdx.x & 31;
    const int w    = (blockIdx.x * blockDim.x + threadIdx.x) >> 5;
    const int tn   = w & 15;       // N/64 = 16 tiles
    const int tm   = w >> 4;       // M/32 tiles
    const int m    = lane & 15;
    const int kb   = (lane < 16) ? 0 : 8;   // per-half-wave K sub-block

    const __bf16* arow0 = A + (size_t)(tm * 32 + m) * EMB;
    const __bf16* arow1 = arow0 + (size_t)16 * EMB;
    const __bf16* brow[4];
#pragma unroll
    for (int nj = 0; nj < 4; ++nj)
        brow[nj] = Bt + (size_t)(tn * 64 + nj * 16 + m) * EMB;

    v8f acc[2][4];
#pragma unroll
    for (int mi = 0; mi < 2; ++mi)
#pragma unroll
        for (int nj = 0; nj < 4; ++nj) acc[mi][nj] = (v8f){};

#pragma unroll 1
    for (int kc = 0; kc < EMB; kc += 32) {
        __builtin_prefetch(arow0 + kc + 256, 0, 1);   // global_prefetch_b8
        __builtin_prefetch(arow1 + kc + 256, 0, 1);
        v16bf a0 = mk16(ld8(arow0 + kc + kb), ld8(arow0 + kc + kb + 16));
        v16bf a1 = mk16(ld8(arow1 + kc + kb), ld8(arow1 + kc + kb + 16));
#pragma unroll
        for (int nj = 0; nj < 4; ++nj) {
            v16bf b = mk16(ld8(brow[nj] + kc + kb), ld8(brow[nj] + kc + kb + 16));
            acc[0][nj] = wmma_bf16(a0, b, acc[0][nj]);
            acc[1][nj] = wmma_bf16(a1, b, acc[1][nj]);
        }
    }

    const int mh = (lane >> 4) << 3;   // C layout: lanes 16-31 hold M = r+8
#pragma unroll
    for (int mi = 0; mi < 2; ++mi) {
#pragma unroll
        for (int nj = 0; nj < 4; ++nj) {
#pragma unroll
            for (int r = 0; r < 8; ++r) {
                int row = tm * 32 + mi * 16 + mh + r;
                int col = tn * 64 + nj * 16 + m;
                float v = acc[mi][nj][r] + bias[col];
                if (mode == 3) {
                    ((float*)outp)[(size_t)row * EMB + col] = v;
                } else {
                    int b_ = row >> 12, s = row & (SEQ - 1);
                    int h  = col >> 6,  d = col & (HD - 1);
                    size_t bh = (size_t)(b_ * NH + h);
                    size_t idx = (mode == 1) ? ((bh * SEQ + s) * HD + d)
                                             : ((bh * HD + d) * SEQ + s);
                    ((__bf16*)outp)[idx] = (__bf16)v;
                }
            }
        }
    }
}

// ---------------------------------------------------------------------------
// Stage 2: windowed attention. One wave per (b,h,16-query tile).
//  Q,K : [BH, S, D] bf16      Vt : [BH, D, S] bf16
//  out : attn [B, S, E] bf16
// ---------------------------------------------------------------------------
__global__ void __launch_bounds__(64)
k_attn(const __bf16* __restrict__ Q, const __bf16* __restrict__ K,
       const __bf16* __restrict__ Vt, __bf16* __restrict__ attn) {
    __shared__ __align__(16) float  sc[2][16][NKPAD];   // f32 scores / exp
    __shared__ __align__(16) __bf16 pb[2][16][NKPAD];   // bf16 probabilities

    const int lane = threadIdx.x & 31;
    const int wv   = threadIdx.x >> 5;
    const int gw   = blockIdx.x * 2 + wv;    // 0 .. B*H*(S/16)-1
    const int qt   = gw & 255;               // S/16 = 256 query tiles
    const int bh   = gw >> 8;
    const int q0   = qt << 4;
    const int m    = lane & 15;
    const int kb   = (lane < 16) ? 0 : 8;
    const int key0 = q0 - WIN;

    // Q tile as two 16x32 A operands (D split 0..31 / 32..63)
    const __bf16* qrow = Q + ((size_t)bh * SEQ + (q0 + m)) * HD;
    v16bf aq0 = mk16(ld8(qrow + kb),      ld8(qrow + kb + 16));
    v16bf aq1 = mk16(ld8(qrow + 32 + kb), ld8(qrow + 32 + kb + 16));

    // ---- scores: 17 key tiles of 16, 2 WMMAs each ----
    const __bf16* Kbase = K + (size_t)bh * SEQ * HD;
#pragma unroll 1
    for (int kt = 0; kt < 17; ++kt) {
        int key = key0 + kt * 16 + m;        // this lane's key column
        v16bf b0 = {}, b1 = {};
        if (key >= 0 && key < SEQ) {
            const __bf16* krow = Kbase + (size_t)key * HD;
            b0 = mk16(ld8(krow + kb),      ld8(krow + kb + 16));
            b1 = mk16(ld8(krow + 32 + kb), ld8(krow + 32 + kb + 16));
        }
        v8f c = {};
        c = wmma_bf16(aq0, b0, c);
        c = wmma_bf16(aq1, b1, c);
#pragma unroll
        for (int r = 0; r < 8; ++r) {
            int mm = r + ((lane >> 4) << 3);
            int q  = q0 + mm;
            bool valid = (key >= 0) && (key < SEQ) &&
                         (key >= q - WIN) && (key <= q + WIN);
            sc[wv][mm][kt * 16 + m] = valid ? c[r] * 0.125f : -3.0e38f;
        }
    }
    __syncthreads();

    // ---- softmax: 2 lanes per row, 136 columns each ----
    {
        const int row = lane >> 1;
        const int seg = lane & 1;
        float* srow = &sc[wv][row][seg * 136];
        float mx = -3.0e38f;
#pragma unroll 8
        for (int j = 0; j < 136; ++j) mx = fmaxf(mx, srow[j]);
        mx = fmaxf(mx, __shfl_xor(mx, 1));
        float s = 0.0f;
#pragma unroll 8
        for (int j = 0; j < 136; ++j) {
            float e = __expf(srow[j] - mx);
            srow[j] = e;
            s += e;
        }
        s += __shfl_xor(s, 1);
        float inv = 1.0f / s;
        __bf16* prow = &pb[wv][row][seg * 136];
#pragma unroll 8
        for (int j = 0; j < 136; ++j) prow[j] = (__bf16)(srow[j] * inv);
        __bf16* ptail = &pb[wv][row][NKEY + seg * 8];
#pragma unroll
        for (int j = 0; j < 8; ++j) ptail[j] = (__bf16)0.0f;   // pad cols 272..287
    }
    __syncthreads();

    // ---- PV: P(16x288 bf16 from LDS) x V(288x64) via 9x4 WMMAs.
    //      One LDS A tile per k-chunk feeds all 4 D-chunk accumulators. ----
    const __bf16* vb = Vt + (size_t)bh * HD * SEQ;
    const int b_ = bh >> 4, h = bh & 15;
    v8f acc[4] = {(v8f){}, (v8f){}, (v8f){}, (v8f){}};
#pragma unroll 1
    for (int kc = 0; kc < 9; ++kc) {
        const __bf16* pr = &pb[wv][m][kc * 32 + kb];
        v16bf a = mk16(ld8(pr), ld8(pr + 16));               // ds_load_b128 x2
        int st = key0 + kc * 32 + kb;
#pragma unroll
        for (int dc = 0; dc < 4; ++dc) {
            const __bf16* vrow = vb + (size_t)(dc * 16 + m) * SEQ;   // B col = D idx
            v16bf b = mk16(ld8_keys(vrow, st, SEQ), ld8_keys(vrow, st + 16, SEQ));
            acc[dc] = wmma_bf16(a, b, acc[dc]);
        }
    }
#pragma unroll
    for (int dc = 0; dc < 4; ++dc) {
#pragma unroll
        for (int r = 0; r < 8; ++r) {
            int mm = r + ((lane >> 4) << 3);
            int q  = q0 + mm;
            int col = h * HD + dc * 16 + m;
            attn[((size_t)(b_ * SEQ + q)) * EMB + col] = (__bf16)acc[dc][r];
        }
    }
}

// ---------------------------------------------------------------------------
extern "C" void kernel_launch(void* const* d_in, const int* in_sizes, int n_in,
                              void* d_out, int out_size, void* d_ws, size_t ws_size,
                              hipStream_t stream) {
    (void)in_sizes; (void)n_in; (void)out_size; (void)ws_size;

    const float* x  = (const float*)d_in[0];
    const float* Wq = (const float*)d_in[1];
    const float* bq = (const float*)d_in[2];
    const float* Wk = (const float*)d_in[3];
    const float* bk = (const float*)d_in[4];
    const float* Wv = (const float*)d_in[5];
    const float* bv = (const float*)d_in[6];
    const float* Wo = (const float*)d_in[7];
    const float* bo = (const float*)d_in[8];
    float* out = (float*)d_out;

    const size_t M  = 4 * (size_t)SEQ;    // 16384 rows
    const size_t NE = M * EMB;            // 16.7M elements
    const size_t WE = (size_t)EMB * EMB;  // 1M elements

    char* p = (char*)d_ws;
    __bf16* xb   = (__bf16*)p; p += NE * 2;
    __bf16* wqT  = (__bf16*)p; p += WE * 2;
    __bf16* wkT  = (__bf16*)p; p += WE * 2;
    __bf16* wvT  = (__bf16*)p; p += WE * 2;
    __bf16* woT  = (__bf16*)p; p += WE * 2;
    __bf16* Qb   = (__bf16*)p; p += NE * 2;
    __bf16* Kb   = (__bf16*)p; p += NE * 2;
    __bf16* Vt   = (__bf16*)p; p += NE * 2;
    __bf16* attn = (__bf16*)p; p += NE * 2;   // total ~168 MB workspace

    // Stage 0: conversions
    k_cvt_bf16<<<(int)(NE / 256), 256, 0, stream>>>(x, xb, (int)NE);
    k_cvt_transpose<<<(int)(WE / 256), 256, 0, stream>>>(Wq, wqT);
    k_cvt_transpose<<<(int)(WE / 256), 256, 0, stream>>>(Wk, wkT);
    k_cvt_transpose<<<(int)(WE / 256), 256, 0, stream>>>(Wv, wvT);
    k_cvt_transpose<<<(int)(WE / 256), 256, 0, stream>>>(Wo, woT);

    // Stage 1: QKV projections (one wave per 32x64 tile, 8 waves/block)
    const int gemmGrid = (int)((M / 32) * (EMB / 64) * 32 / 256);   // 1024
    k_gemm<<<gemmGrid, 256, 0, stream>>>(xb, wqT, bq, Qb, 1);
    k_gemm<<<gemmGrid, 256, 0, stream>>>(xb, wkT, bk, Kb, 1);
    k_gemm<<<gemmGrid, 256, 0, stream>>>(xb, wvT, bv, Vt, 2);

    // Stage 2: windowed attention (16384 tile-waves, 2 waves/block)
    const int attnGrid = (int)(4 * NH * (SEQ / 16) / 2);            // 8192
    k_attn<<<attnGrid, 64, 0, stream>>>(Qb, Kb, Vt, attn);

    // Stage 3: output projection, fp32 result + bias
    k_gemm<<<gemmGrid, 256, 0, stream>>>(attn, woT, bo, (void*)out, 3);
}
